// MGCGRU_13408887898586
// MI455X (gfx1250) — matrix operand order repaired and checked by reading, not compiled, not run on hardware
//
#include <hip/hip_runtime.h>
#include <hip/hip_bf16.h>
#include <stdint.h>

typedef __attribute__((ext_vector_type(16))) _Float16 v16h;
typedef __attribute__((ext_vector_type(8)))  _Float16 v8h;
typedef __attribute__((ext_vector_type(8)))  float    v8f;
typedef __attribute__((ext_vector_type(4)))  unsigned int v4u;
typedef __attribute__((ext_vector_type(8)))  int      v8i_t;
typedef __attribute__((ext_vector_type(4)))  int      v4i_t;

#define B_    8192
#define V_    101
#define T_    30
#define C_    4
#define DIM_  64
#define HID_  128
#define MP    112      // V padded to 7*16 (M dimension)
#define KPV   128      // V padded to 128 (K dimension)
#define KPC   32       // C padded to 32 (K dimension)
#define FLAT  6464     // V_*DIM_

// ---------------- workspace layout (units: _Float16 elements) ----------------
static constexpr size_t OFF_DYN  = 0;                                   // MP*KPV
static constexpr size_t OFF_W1T  = OFF_DYN  + (size_t)MP  * KPV;        // DIM*KPC
static constexpr size_t OFF_W2T  = OFF_W1T  + (size_t)DIM_* KPC;        // DIM*DIM
static constexpr size_t OFF_WZX  = OFF_W2T  + (size_t)DIM_* DIM_;
static constexpr size_t OFF_WZF  = OFF_WZX  + (size_t)DIM_* KPC;
static constexpr size_t OFF_WZH  = OFF_WZF  + (size_t)DIM_* DIM_;
static constexpr size_t OFF_WRX  = OFF_WZH  + (size_t)DIM_* DIM_;
static constexpr size_t OFF_WRF  = OFF_WRX  + (size_t)DIM_* KPC;
static constexpr size_t OFF_WRH  = OFF_WRF  + (size_t)DIM_* DIM_;
static constexpr size_t OFF_WWX  = OFF_WRH  + (size_t)DIM_* DIM_;
static constexpr size_t OFF_WWF  = OFF_WWX  + (size_t)DIM_* KPC;
static constexpr size_t OFF_WWH  = OFF_WWF  + (size_t)DIM_* DIM_;
static constexpr size_t OFF_WF1T = OFF_WWH  + (size_t)DIM_* DIM_;       // HID*FLAT
static constexpr size_t OFF_HFIN = OFF_WF1T + (size_t)HID_* FLAT;       // B_*FLAT

#if defined(__HIP_DEVICE_COMPILE__) && __has_builtin(__builtin_amdgcn_tensor_load_to_lds)
#define USE_TDM 1
#else
#define USE_TDM 0
#endif

// ---------------- WMMA fragment helpers (ISA 7.12.2 layouts) ----------------
// A fragment, row-major A with leading dim lda (halfs). Lane L<16: M=L, K=e<8?e:16+e-8
// high half-wave adds +8 to K base. Two contiguous 16B loads per lane.
__device__ __forceinline__ v16h load_a_frag(const _Float16* A, int lda, int k0, int lane) {
    const _Float16* p = A + (lane & 15) * lda + k0 + ((lane >> 4) << 3);
    v8h lo = *(const v8h*)p;
    v8h hi = *(const v8h*)(p + 16);
    return __builtin_shufflevector(lo, hi, 0,1,2,3,4,5,6,7,8,9,10,11,12,13,14,15);
}

// B fragment from a transposed (N-major) buffer Bt[N][K], ldk halfs.
// Lane L: N = L&15, K = (L>>4)*8 + (e<8?e:16+e-8)  -> two contiguous 16B runs per lane.
__device__ __forceinline__ v16h load_bt_frag(const _Float16* Bt, int ldk, int k0, int lane) {
    const _Float16* p = Bt + (lane & 15) * ldk + k0 + ((lane >> 4) << 3);
    v8h lo = *(const v8h*)p;
    v8h hi = *(const v8h*)(p + 16);
    return __builtin_shufflevector(lo, hi, 0,1,2,3,4,5,6,7,8,9,10,11,12,13,14,15);
}

__device__ __forceinline__ v8f gemm_acc(v8f acc, const _Float16* A, int lda,
                                        const _Float16* Bt, int ldk, int K, int lane) {
    for (int k = 0; k < K; k += 32) {
        v16h a = load_a_frag(A, lda, k, lane);
        v16h b = load_bt_frag(Bt, ldk, k, lane);
        acc = __builtin_amdgcn_wmma_f32_16x16x32_f16(false, a, false, b,
                                                     (short)0, acc, false, false);
    }
    return acc;
}

__device__ __forceinline__ v8f bcast8(float v) {
    v8f a; for (int r = 0; r < 8; ++r) a[r] = v; return a;
}

// C-tile (16x16) store, transposed into Bt[N][K] : one contiguous b128 per lane
__device__ __forceinline__ void store_c_trans(v8f acc, _Float16* Bt, int ldk,
                                              int n0, int m0, int lane) {
    _Float16* p = Bt + (n0 + (lane & 15)) * ldk + m0 + ((lane >> 4) << 3);
    v8h o; for (int r = 0; r < 8; ++r) o[r] = (_Float16)acc[r];
    *(v8h*)p = o;
}

// ---------------- weight prep: fp32 -> padded/transposed fp16 in ws ----------------
__global__ __launch_bounds__(256) void prep_kernel(
    const float* __restrict__ dyn, const float* __restrict__ W1,
    const float* __restrict__ W2,  const float* __restrict__ Wz,
    const float* __restrict__ Wr,  const float* __restrict__ Ww,
    const float* __restrict__ Wf1, _Float16* __restrict__ ws)
{
    const int tid = blockIdx.x * blockDim.x + threadIdx.x;
    const int nt  = gridDim.x * blockDim.x;

    // dyn: row-major, zero padded to MP x KPV (A-role)
    for (int i = tid; i < MP * KPV; i += nt) {
        int u = i / KPV, v = i % KPV;
        ws[OFF_DYN + i] = (u < V_ && v < V_) ? (_Float16)dyn[u * V_ + v] : (_Float16)0.f;
    }
    // transposed weights Wt[n][k] (B-role), zero-padded K
    auto fillT = [&](size_t off, const float* src, int K, int Kp, int rowOff) {
        for (int i = tid; i < DIM_ * Kp; i += nt) {
            int n = i / Kp, k = i % Kp;
            ws[off + i] = (k < K) ? (_Float16)src[(size_t)(rowOff + k) * DIM_ + n]
                                  : (_Float16)0.f;
        }
    };
    fillT(OFF_W1T, W1, C_,   KPC,  0);
    fillT(OFF_W2T, W2, DIM_, DIM_, 0);
    fillT(OFF_WZX, Wz, C_,   KPC,  0);
    fillT(OFF_WZF, Wz, DIM_, DIM_, C_);
    fillT(OFF_WZH, Wz, DIM_, DIM_, C_ + DIM_);
    fillT(OFF_WRX, Wr, C_,   KPC,  0);
    fillT(OFF_WRF, Wr, DIM_, DIM_, C_);
    fillT(OFF_WRH, Wr, DIM_, DIM_, C_ + DIM_);
    fillT(OFF_WWX, Ww, C_,   KPC,  0);
    fillT(OFF_WWF, Ww, DIM_, DIM_, C_);
    fillT(OFF_WWH, Ww, DIM_, DIM_, C_ + DIM_);
    // Wf1 (FLAT x HID) -> Wf1t[HID][FLAT]
    for (size_t i = tid; i < (size_t)HID_ * FLAT; i += nt) {
        size_t n = i / FLAT, k = i % FLAT;
        ws[OFF_WF1T + i] = (_Float16)Wf1[k * HID_ + n];
    }
}

// ---------------- GRU recurrence: one batch element per workgroup ----------------
__global__ __launch_bounds__(128) void recurrence_kernel(
    const float* __restrict__ x,  const float* __restrict__ b1,
    const float* __restrict__ b2, const float* __restrict__ bz,
    const float* __restrict__ br, const float* __restrict__ bw,
    const _Float16* __restrict__ ws, _Float16* __restrict__ hfin)
{
    __shared__ __align__(16) _Float16 sDyn[MP * KPV];   // dyn, A-role
    __shared__ __align__(16) _Float16 sXt [MP * KPC];   // xt padded, A-role
    __shared__ __align__(16) _Float16 sBt [DIM_ * KPV]; // u^T / v^T, B-role
    __shared__ __align__(16) _Float16 sF  [MP * DIM_];  // f, A-role
    __shared__ __align__(16) _Float16 sH  [MP * DIM_];  // h, A-role + elementwise
    __shared__ __align__(16) _Float16 sRH [MP * DIM_];  // r*h, A-role

    const int b    = blockIdx.x;
    const int tid  = threadIdx.x;
    const int lane = tid & 31;
    const int wave = tid >> 5;
    const int n0   = wave << 4;          // each wave owns a 16-wide N strip
    const int ln   = lane & 15;
    const int hh   = lane >> 4;

    // one-time zero fills (pad regions are never touched again -> stay zero)
    for (int i = tid; i < MP * KPC;   i += 128) sXt[i] = (_Float16)0.f;
    for (int i = tid; i < DIM_ * KPV; i += 128) sBt[i] = (_Float16)0.f;
    for (int i = tid; i < MP * DIM_;  i += 128) sH[i]  = (_Float16)0.f;

#if USE_TDM
    if (tid < 32) {
        // Tensor Data Mover: copy padded dyn (MP x KPV fp16) ws->LDS, 2D tile
        unsigned long long ga = (unsigned long long)(uintptr_t)(const void*)(ws + OFF_DYN);
        unsigned lds_off = (unsigned)(unsigned long long)(uintptr_t)(void*)&sDyn[0];
        v4u g0;
        g0.x = 1u;                                    // count=1, user descriptor
        g0.y = lds_off;                               // lds_addr
        g0.z = (unsigned)(ga & 0xffffffffull);        // global_addr[31:0]
        g0.w = (unsigned)((ga >> 32) & 0x01ffffffull) | (2u << 30); // addr hi | type=2
        v8i_t g1;
        g1[0] = (1 << 16);           // data_size = 2 bytes
        g1[1] = (KPV << 16);         // tensor_dim0 low 16 (in bits 79:48)
        g1[2] = (MP  << 16);         // tensor_dim1 low 16
        g1[3] = (KPV << 16);         // tile_dim0
        g1[4] = MP;                  // tile_dim1 (tile_dim2 = 0)
        g1[5] = KPV;                 // tensor_dim0_stride low 32
        g1[6] = 0;
        g1[7] = 0;
        v4i_t g2 = {0, 0, 0, 0};     // group 2 unused (2D tensor)
        v4i_t g3 = {0, 0, 0, 0};     // group 3 unused
        v8i_t g4 = {0, 0, 0, 0, 0, 0, 0, 0}; // extra group (clang-23 6-arg form)
        __builtin_amdgcn_tensor_load_to_lds(g0, g1, g2, g3, g4, 0);
    }
    __builtin_amdgcn_s_wait_tensorcnt(0);
#else
    for (int i = tid; i < MP * KPV; i += 128) sDyn[i] = ws[OFF_DYN + i];
#endif
    __syncthreads();

    const float* xb = x + (size_t)b * V_ * T_ * C_;

    for (int t = 0; t < T_; ++t) {
        __syncthreads();
        // ---- load x_t (V_ x C_ fp32, strided) into padded fp16 tile
        for (int i = tid; i < V_ * C_; i += 128) {
            int v = i >> 2, c = i & 3;
            sXt[v * KPC + c] = (_Float16)xb[(v * T_ + t) * C_ + c];
        }
        if (t + 1 < T_ && tid < V_)
            __builtin_prefetch(&xb[(tid * T_ + t + 1) * C_], 0, 1);
        __syncthreads();

        // ---- u^T = (xt @ W1)^T   (dyn@(xt@W1) == (dyn@xt)@W1)
        for (int mt = 0; mt < 7; ++mt) {
            v8f acc = bcast8(0.f);
            acc = gemm_acc(acc, sXt + mt * 16 * KPC, KPC,
                           ws + OFF_W1T + (size_t)n0 * KPC, KPC, KPC, lane);
            store_c_trans(acc, sBt, KPV, n0, mt * 16, lane);
        }
        __syncthreads();

        // ---- f = tanh(dyn @ u + b1)
        {
            float bb = b1[n0 + ln];
            for (int mt = 0; mt < 7; ++mt) {
                v8f acc = bcast8(bb);
                acc = gemm_acc(acc, sDyn + mt * 16 * KPV, KPV,
                               sBt + (size_t)n0 * KPV, KPV, KPV, lane);
                for (int r = 0; r < 8; ++r) {
                    int m = mt * 16 + (hh << 3) + r;
                    sF[m * DIM_ + n0 + ln] = (_Float16)tanhf(acc[r]);
                }
            }
        }
        __syncthreads();

        // ---- v^T = (f @ W2)^T
        for (int mt = 0; mt < 7; ++mt) {
            v8f acc = bcast8(0.f);
            acc = gemm_acc(acc, sF + mt * 16 * DIM_, DIM_,
                           ws + OFF_W2T + (size_t)n0 * DIM_, DIM_, DIM_, lane);
            store_c_trans(acc, sBt, KPV, n0, mt * 16, lane);
        }
        __syncthreads();

        // ---- f = tanh(dyn @ v + b2)
        {
            float bb = b2[n0 + ln];
            for (int mt = 0; mt < 7; ++mt) {
                v8f acc = bcast8(bb);
                acc = gemm_acc(acc, sDyn + mt * 16 * KPV, KPV,
                               sBt + (size_t)n0 * KPV, KPV, KPV, lane);
                for (int r = 0; r < 8; ++r) {
                    int m = mt * 16 + (hh << 3) + r;
                    sF[m * DIM_ + n0 + ln] = (_Float16)tanhf(acc[r]);
                }
            }
        }
        __syncthreads();

        // ---- gates: z (kept in regs), r -> rh in LDS
        v8f zsave[7];
        {
            float bbz = bz[n0 + ln], bbr = br[n0 + ln];
            for (int mt = 0; mt < 7; ++mt) {
                const _Float16* Ax = sXt + mt * 16 * KPC;
                const _Float16* Af = sF  + mt * 16 * DIM_;
                const _Float16* Ah = sH  + mt * 16 * DIM_;
                v8f za = bcast8(bbz);
                za = gemm_acc(za, Ax, KPC,  ws + OFF_WZX + (size_t)n0 * KPC,  KPC,  KPC,  lane);
                za = gemm_acc(za, Af, DIM_, ws + OFF_WZF + (size_t)n0 * DIM_, DIM_, DIM_, lane);
                za = gemm_acc(za, Ah, DIM_, ws + OFF_WZH + (size_t)n0 * DIM_, DIM_, DIM_, lane);
                v8f ra = bcast8(bbr);
                ra = gemm_acc(ra, Ax, KPC,  ws + OFF_WRX + (size_t)n0 * KPC,  KPC,  KPC,  lane);
                ra = gemm_acc(ra, Af, DIM_, ws + OFF_WRF + (size_t)n0 * DIM_, DIM_, DIM_, lane);
                ra = gemm_acc(ra, Ah, DIM_, ws + OFF_WRH + (size_t)n0 * DIM_, DIM_, DIM_, lane);
                for (int r = 0; r < 8; ++r) {
                    float zg = 1.f / (1.f + __expf(-za[r]));
                    float rg = 1.f / (1.f + __expf(-ra[r]));
                    za[r] = zg;
                    int m = mt * 16 + (hh << 3) + r;
                    int idx = m * DIM_ + n0 + ln;
                    sRH[idx] = (_Float16)(rg * (float)sH[idx]);
                }
                zsave[mt] = za;
            }
        }
        __syncthreads();

        // ---- cn = tanh(...), h = (1-z)*h + z*cn
        {
            float bbw = bw[n0 + ln];
            for (int mt = 0; mt < 7; ++mt) {
                v8f acc = bcast8(bbw);
                acc = gemm_acc(acc, sXt + mt * 16 * KPC,  KPC,
                               ws + OFF_WWX + (size_t)n0 * KPC,  KPC,  KPC,  lane);
                acc = gemm_acc(acc, sF  + mt * 16 * DIM_, DIM_,
                               ws + OFF_WWF + (size_t)n0 * DIM_, DIM_, DIM_, lane);
                acc = gemm_acc(acc, sRH + mt * 16 * DIM_, DIM_,
                               ws + OFF_WWH + (size_t)n0 * DIM_, DIM_, DIM_, lane);
                for (int r = 0; r < 8; ++r) {
                    float cn = tanhf(acc[r]);
                    int m = mt * 16 + (hh << 3) + r;
                    int idx = m * DIM_ + n0 + ln;
                    float hv = (float)sH[idx];
                    float zg = zsave[mt][r];
                    sH[idx] = (_Float16)((1.f - zg) * hv + zg * cn);
                }
            }
        }
    }
    __syncthreads();

    // ---- flush final h (valid nodes only) to global fp16
    _Float16* ho = hfin + (size_t)b * FLAT;
    for (int i = tid; i < FLAT; i += 128) ho[i] = sH[i];
}

// ---------------- final MLP + softmax: 16 batch rows per block ----------------
__global__ __launch_bounds__(256) void mlp_kernel(
    const _Float16* __restrict__ hfin, const _Float16* __restrict__ wf1t,
    const float* __restrict__ bf1,     const float* __restrict__ wf2,
    const float* __restrict__ bf2,     float* __restrict__ out)
{
    __shared__ __align__(16) float sAct[16 * HID_];
    __shared__ float sLog[16 * 112];

    const int tid  = threadIdx.x;
    const int lane = tid & 31;
    const int wave = tid >> 5;      // 8 waves -> N strips of 16 over HID=128
    const int n0   = wave << 4;
    const int ln   = lane & 15;
    const int hh   = lane >> 4;
    const int row0 = blockIdx.x * 16;

    // GEMM1: (16 x FLAT) @ (FLAT x 128), relu fused
    const _Float16* A = hfin + (size_t)row0 * FLAT;
    v8f acc = bcast8(bf1[n0 + ln]);
    acc = gemm_acc(acc, A, FLAT, wf1t + (size_t)n0 * FLAT, FLAT, FLAT, lane);
    for (int r = 0; r < 8; ++r) {
        int m = (hh << 3) + r;
        sAct[m * HID_ + n0 + ln] = fmaxf(acc[r], 0.f);
    }
    __syncthreads();

    // GEMM2: (16 x 128) @ (128 x 101), one output column per thread
    if (tid < V_) {
        for (int m = 0; m < 16; ++m) {
            float s = bf2[tid];
            for (int k = 0; k < HID_; ++k)
                s += sAct[m * HID_ + k] * wf2[k * V_ + tid];
            sLog[m * 112 + tid] = s;
        }
    }
    __syncthreads();

    // softmax, one row per thread
    if (tid < 16) {
        float mx = -1e30f;
        for (int n = 0; n < V_; ++n) mx = fmaxf(mx, sLog[tid * 112 + n]);
        float sum = 0.f;
        for (int n = 0; n < V_; ++n) sum += __expf(sLog[tid * 112 + n] - mx);
        float inv = 1.f / sum;
        for (int n = 0; n < V_; ++n)
            out[(size_t)(row0 + tid) * V_ + n] = __expf(sLog[tid * 112 + n] - mx) * inv;
    }
}

// ---------------- launch ----------------
extern "C" void kernel_launch(void* const* d_in, const int* in_sizes, int n_in,
                              void* d_out, int out_size, void* d_ws, size_t ws_size,
                              hipStream_t stream) {
    const float* x   = (const float*)d_in[0];
    const float* dyn = (const float*)d_in[1];
    const float* W1  = (const float*)d_in[2];
    const float* b1  = (const float*)d_in[3];
    const float* W2  = (const float*)d_in[4];
    const float* b2  = (const float*)d_in[5];
    const float* Wz  = (const float*)d_in[6];
    const float* bz  = (const float*)d_in[7];
    const float* Wr  = (const float*)d_in[8];
    const float* br  = (const float*)d_in[9];
    const float* Ww  = (const float*)d_in[10];
    const float* bw  = (const float*)d_in[11];
    const float* Wf1 = (const float*)d_in[12];
    const float* bf1 = (const float*)d_in[13];
    const float* Wf2 = (const float*)d_in[14];
    const float* bf2 = (const float*)d_in[15];
    float* out = (float*)d_out;
    _Float16* ws = (_Float16*)d_ws;

    prep_kernel<<<256, 256, 0, stream>>>(dyn, W1, W2, Wz, Wr, Ww, Wf1, ws);
    recurrence_kernel<<<B_, 128, 0, stream>>>(x, b1, b2, bz, br, bw, ws,
                                              ws + OFF_HFIN);
    mlp_kernel<<<B_ / 16, 256, 0, stream>>>(ws + OFF_HFIN, ws + OFF_WF1T,
                                            bf1, Wf2, bf2, out);
}
